// GPT_15436112462648
// MI455X (gfx1250) — matrix operand order repaired and checked by reading, not compiled, not run on hardware
//
#include <hip/hip_runtime.h>
#include <hip/hip_bf16.h>
#include <math.h>

typedef __attribute__((ext_vector_type(16))) _Float16 v16h;
typedef __attribute__((ext_vector_type(8)))  _Float16 v8h;
typedef __attribute__((ext_vector_type(8)))  float    v8f;

#define Dm 768
#define Fm 3072
#define Vm 50257
#define Lm 12
#define Sm 1024
#define Bm 2
#define Mm (Bm * Sm)   // 2048 rows

#define BM 128
#define BN 128
#define BK 32
#define LDA 40   // sA row stride (halves): 80B, 16B-aligned, 20-bank stride
#define LDB 40   // sB (column-major) row stride

__device__ __forceinline__ v8f wmma_f16(v16h a, v16h b, v8f c) {
  return __builtin_amdgcn_wmma_f32_16x16x32_f16(false, a, false, b, (short)0, c, false, false);
}

__device__ __forceinline__ v16h cat8(v8h lo, v8h hi) {
  return __builtin_shufflevector(lo, hi, 0, 1, 2, 3, 4, 5, 6, 7, 8, 9, 10, 11, 12, 13, 14, 15);
}

// 8 consecutive f32 -> v8h (two global_load_b128 + pack)
__device__ __forceinline__ v8h cvt8(const float* p) {
  const float4 a = *(const float4*)p;
  const float4 b = *(const float4*)(p + 4);
  v8h r;
  r[0] = (_Float16)a.x; r[1] = (_Float16)a.y; r[2] = (_Float16)a.z; r[3] = (_Float16)a.w;
  r[4] = (_Float16)b.x; r[5] = (_Float16)b.y; r[6] = (_Float16)b.z; r[7] = (_Float16)b.w;
  return r;
}

// A-fragment 16x32 from row-major f16 LDS (ISA A layout: half0 K=0..7,16..23;
// half1 K=8..15,24..31). ld multiple of 8 halves; 16B-aligned runs.
__device__ __forceinline__ v16h load_fragA(const _Float16* p, int ld) {
  const int lane = threadIdx.x & 31;
  const _Float16* q = p + (lane & 15) * ld + ((lane >> 4) << 3);
  const v8h lo = *(const v8h*)q;
  const v8h hi = *(const v8h*)(q + 16);
  return cat8(lo, hi);
}

// B-fragment 32x16 from column-major f16 LDS tile colmaj[col][k]
// (half0 K=0..7,16..23; half1 K=8..15,24..31) -> two 16B runs per lane.
__device__ __forceinline__ v16h load_fragB_cm(const _Float16* p, int ld) {
  const int lane = threadIdx.x & 31;
  const _Float16* q = p + (lane & 15) * ld + ((lane >> 4) << 3);
  const v8h lo = *(const v8h*)q;
  const v8h hi = *(const v8h*)(q + 16);
  return cat8(lo, hi);
}

// -------------------- embedding + positional encoding --------------------
__global__ __launch_bounds__(256) void embed_k(const float* __restrict__ emb,
                                               const int* __restrict__ x,
                                               float* __restrict__ H) {
  const int rs = blockIdx.x;
  const int s = rs & (Sm - 1);
  const int tok = x[rs];
#pragma unroll
  for (int j = 0; j < 3; ++j) {
    const int c = threadIdx.x + j * 256;
    const float freq = __expf(-(float)(c & ~1) * (9.210340371976184f / (float)Dm));
    const float ang = (float)s * freq;
    const float pe = (c & 1) ? __cosf(ang) : __sinf(ang);
    H[(size_t)rs * Dm + c] = emb[(size_t)tok * Dm + c] * 27.712812921102035f + pe;
  }
}

// -------------------- layernorm (ddof=1 std, eps added to std) -----------
__global__ __launch_bounds__(256) void layernorm_k(const float* __restrict__ X,
                                                   const float* __restrict__ ga,
                                                   const float* __restrict__ gb,
                                                   float* __restrict__ Y) {
  __shared__ float s1[256], s2[256];
  const int row = blockIdx.x, tid = threadIdx.x;
  const float* xr = X + (size_t)row * Dm;
  float xs[3];
  float sum = 0.f, ssq = 0.f;
#pragma unroll
  for (int j = 0; j < 3; ++j) {
    float v = xr[tid + j * 256];
    xs[j] = v; sum += v; ssq += v * v;
  }
  s1[tid] = sum; s2[tid] = ssq;
  __syncthreads();
  for (int s = 128; s > 0; s >>= 1) {
    if (tid < s) { s1[tid] += s1[tid + s]; s2[tid] += s2[tid + s]; }
    __syncthreads();
  }
  const float mean = s1[0] / (float)Dm;
  const float var = fmaxf(s2[0] - (float)Dm * mean * mean, 0.f) / (float)(Dm - 1);
  const float inv = 1.f / (sqrtf(var) + 1e-6f);
#pragma unroll
  for (int j = 0; j < 3; ++j) {
    const int c = tid + j * 256;
    Y[(size_t)row * Dm + c] = ga[c] * (xs[j] - mean) * inv + gb[c];
  }
}

// -------------------- generic WMMA GEMM ----------------------------------
// C[M,N] = act(A[M,K] * B + bias + res) ; transB: B stored [N,K] row-major
// 128x128 block, 8 waves, each wave owns 32x64 (2x4 WMMA tiles -> 8 WMMA/step)
__global__ __launch_bounds__(256) void wmma_gemm(
    const float* __restrict__ A, const float* __restrict__ B,
    const float* __restrict__ bias, const float* res, float* C,
    int M, int N, int K, int transB, int doRelu) {
  __shared__ __align__(32) _Float16 sA[BM][LDA];
  __shared__ __align__(32) _Float16 sB[BN][LDB];   // column-major: sB[col][k]
  const int tid = threadIdx.x;
  const int lane = tid & 31;
  const int wave = tid >> 5;
  const int wm = wave >> 1;           // 0..3 (32-row strip)
  const int wn = wave & 1;            // 0..1 (64-col strip)
  const int m0 = blockIdx.y * BM;
  const int n0 = blockIdx.x * BN;

  v8f acc[2][4];
#pragma unroll
  for (int i = 0; i < 2; ++i)
#pragma unroll
    for (int j = 0; j < 4; ++j) acc[i][j] = (v8f){};

  const int arow = tid >> 1, akoff = (tid & 1) * 16;
  const int bp = tid >> 7, bc = tid & 127;   // bp: k half (16 rows), bc: column
  const bool bok = (n0 + bc) < N;

  for (int k0 = 0; k0 < K; k0 += BK) {
    // A tile: 16 contiguous f32 -> two v8h stores
    {
      const float* ap = &A[(size_t)(m0 + arow) * K + (k0 + akoff)];
      *(v8h*)&sA[arow][akoff] = cvt8(ap);
      *(v8h*)&sA[arow][akoff + 8] = cvt8(ap + 8);
    }
    // B tile, column-major
    v8h x0, x1;
    if (!transB) {
      if (bok) {
#pragma unroll
        for (int j = 0; j < 8; ++j) {
          x0[j] = (_Float16)B[(size_t)(k0 + 16 * bp + j) * N + (n0 + bc)];
          x1[j] = (_Float16)B[(size_t)(k0 + 16 * bp + 8 + j) * N + (n0 + bc)];
        }
      } else { x0 = (v8h){}; x1 = (v8h){}; }
    } else {
      if (bok) {
        const float* bp0 = &B[(size_t)(n0 + bc) * K + (k0 + 16 * bp)];
        x0 = cvt8(bp0);
        x1 = cvt8(bp0 + 8);
      } else { x0 = (v8h){}; x1 = (v8h){}; }
    }
    *(v8h*)&sB[bc][16 * bp] = x0;
    *(v8h*)&sB[bc][16 * bp + 8] = x1;
    __syncthreads();

    const v16h a0 = load_fragA(&sA[wm * 32][0], LDA);
    const v16h a1 = load_fragA(&sA[wm * 32 + 16][0], LDA);
    v16h bfr[4];
#pragma unroll
    for (int j = 0; j < 4; ++j)
      bfr[j] = load_fragB_cm(&sB[wn * 64 + 16 * j][0], LDB);
#pragma unroll
    for (int j = 0; j < 4; ++j) {
      acc[0][j] = wmma_f16(a0, bfr[j], acc[0][j]);
      acc[1][j] = wmma_f16(a1, bfr[j], acc[1][j]);
    }
    __syncthreads();
  }

  const int half = lane >> 4, nlo = lane & 15;
#pragma unroll
  for (int ti = 0; ti < 2; ++ti)
#pragma unroll
    for (int tj = 0; tj < 4; ++tj) {
      const int rb = m0 + wm * 32 + ti * 16;
      const int cb = n0 + wn * 64 + tj * 16 + nlo;
      if (cb < N) {
#pragma unroll
        for (int r = 0; r < 8; ++r) {
          const int row = rb + r + 8 * half;
          float v = acc[ti][tj][r];
          if (bias) v += bias[cb];
          if (res) v += res[(size_t)row * N + cb];
          if (doRelu) v = fmaxf(v, 0.f);
          C[(size_t)row * N + cb] = v;
        }
      }
    }
}

// -------------------- flash attention (1 wave = 16 query rows) -----------
__global__ __launch_bounds__(32) void flash_attn(const float* __restrict__ Q,
                                                 const float* __restrict__ Kg,
                                                 const float* __restrict__ Vg,
                                                 float* __restrict__ O) {
  __shared__ __align__(32) _Float16 sK[32][72];   // [key][dk] (column-major as B)
  __shared__ __align__(32) _Float16 sVt[64][40];  // [dk][key] (column-major as B)
  __shared__ __align__(32) _Float16 sP[16][40];   // [row][key] (row-major as A)
  const int bid = blockIdx.x;
  const int qt = bid & 63;
  const int hh = (bid >> 6) % 12;
  const int bb = bid / (64 * 12);
  const int q0 = qt * 16;
  const int lane = threadIdx.x & 31;
  const int half = lane >> 4, l16 = lane & 15;
  const size_t base = (size_t)bb * Sm * Dm + (size_t)hh * 64;

  // Q fragments (vectorized global loads, interleaved A layout)
  const float* qrow = &Q[base + (size_t)(q0 + l16) * Dm];
  const v16h aq0 = cat8(cvt8(qrow + 8 * half), cvt8(qrow + 16 + 8 * half));
  const v16h aq1 = cat8(cvt8(qrow + 32 + 8 * half), cvt8(qrow + 48 + 8 * half));

  v8f oacc[4];
#pragma unroll
  for (int c = 0; c < 4; ++c) oacc[c] = (v8f){};
  float m[8], lsum[8];
#pragma unroll
  for (int r = 0; r < 8; ++r) { m[r] = -3.0e38f; lsum[r] = 0.f; }

  const int kend = q0 + 16;
  for (int kc = 0; kc < kend; kc += 32) {
    // stage K: lane owns one key row, 64 contiguous dk -> 8 v8h stores
    {
      const float* krow = &Kg[base + (size_t)(kc + lane) * Dm];
#pragma unroll
      for (int j = 0; j < 8; ++j)
        *(v8h*)&sK[lane][j * 8] = cvt8(krow + j * 8);
    }
    // stage V transposed: lane owns dk pair, coalesced float2 over keys
    {
      const int dk0 = lane * 2;
      v8h vb0[4], vb1[4];
#pragma unroll
      for (int j = 0; j < 32; ++j) {
        const float2 vv = *(const float2*)&Vg[base + (size_t)(kc + j) * Dm + dk0];
        vb0[j >> 3][j & 7] = (_Float16)vv.x;
        vb1[j >> 3][j & 7] = (_Float16)vv.y;
      }
#pragma unroll
      for (int s4 = 0; s4 < 4; ++s4) {
        *(v8h*)&sVt[dk0][s4 * 8] = vb0[s4];
        *(v8h*)&sVt[dk0 + 1][s4 * 8] = vb1[s4];
      }
    }
    __syncthreads();

    float sv[2][8];
#pragma unroll
    for (int t = 0; t < 2; ++t) {
      const v16h bkA = load_fragB_cm(&sK[t * 16][0], 72);
      const v16h bkB = load_fragB_cm(&sK[t * 16][32], 72);
      v8f sc = (v8f){};
      sc = wmma_f16(aq0, bkA, sc);
      sc = wmma_f16(aq1, bkB, sc);
#pragma unroll
      for (int r = 0; r < 8; ++r) {
        float v = sc[r] * 0.125f;                    // 1/sqrt(64)
        const int row = q0 + r + 8 * half;
        const int key = kc + t * 16 + l16;
        if (key > row) v = -1e9f;                    // causal mask
        sv[t][r] = v;
      }
    }

    // online softmax update (row stats within owning 16-lane half)
    float p0[8], p1[8];
#pragma unroll
    for (int r = 0; r < 8; ++r) {
      float mx = fmaxf(sv[0][r], sv[1][r]);
      for (int off = 8; off >= 1; off >>= 1) mx = fmaxf(mx, __shfl_xor(mx, off, 32));
      const float mnew = fmaxf(m[r], mx);
      const float scale = __expf(m[r] - mnew);
      p0[r] = __expf(sv[0][r] - mnew);
      p1[r] = __expf(sv[1][r] - mnew);
      float rs = p0[r] + p1[r];
      for (int off = 8; off >= 1; off >>= 1) rs += __shfl_xor(rs, off, 32);
      lsum[r] = lsum[r] * scale + rs;
      m[r] = mnew;
#pragma unroll
      for (int c = 0; c < 4; ++c) oacc[c][r] = oacc[c][r] * scale;
    }

    // relayout P (D-layout) -> A-fragment via LDS
    __syncthreads();
#pragma unroll
    for (int r = 0; r < 8; ++r) {
      const int row = r + 8 * half;
      sP[row][l16] = (_Float16)p0[r];
      sP[row][16 + l16] = (_Float16)p1[r];
    }
    __syncthreads();
    const v16h ap = load_fragA(&sP[0][0], 40);
#pragma unroll
    for (int c = 0; c < 4; ++c) {
      const v16h bv = load_fragB_cm(&sVt[c * 16][0], 40);
      oacc[c] = wmma_f16(ap, bv, oacc[c]);
    }
    __syncthreads();
  }

#pragma unroll
  for (int c = 0; c < 4; ++c)
#pragma unroll
    for (int r = 0; r < 8; ++r) {
      const int row = q0 + r + 8 * half;
      O[base + (size_t)row * Dm + c * 16 + l16] = oacc[c][r] / lsum[r];
    }
}

// -------------------- host orchestration ---------------------------------
extern "C" void kernel_launch(void* const* d_in, const int* in_sizes, int n_in,
                              void* d_out, int out_size, void* d_ws, size_t ws_size,
                              hipStream_t stream) {
  (void)in_sizes; (void)n_in; (void)out_size; (void)ws_size;
  const float* emb   = (const float*)d_in[0];
  const float* Wq    = (const float*)d_in[1];
  const float* Wk    = (const float*)d_in[2];
  const float* Wv    = (const float*)d_in[3];
  const float* Wo    = (const float*)d_in[4];
  const float* ln1_a = (const float*)d_in[5];
  const float* ln1_b = (const float*)d_in[6];
  const float* ln2_a = (const float*)d_in[7];
  const float* ln2_b = (const float*)d_in[8];
  const float* W1    = (const float*)d_in[9];
  const float* b1    = (const float*)d_in[10];
  const float* W2    = (const float*)d_in[11];
  const float* b2    = (const float*)d_in[12];
  const float* fn_a  = (const float*)d_in[13];
  const float* fn_b  = (const float*)d_in[14];
  const float* pb    = (const float*)d_in[15];
  const int*   x     = (const int*)d_in[16];
  float* out = (float*)d_out;

  float* ws = (float*)d_ws;
  float* h  = ws;
  float* xn = h  + (size_t)Mm * Dm;
  float* q  = xn + (size_t)Mm * Dm;
  float* k  = q  + (size_t)Mm * Dm;
  float* v  = k  + (size_t)Mm * Dm;
  float* o  = v  + (size_t)Mm * Dm;
  float* f  = o  + (size_t)Mm * Dm;

  embed_k<<<Mm, 256, 0, stream>>>(emb, x, h);

  const dim3 gDD(Dm / BN, Mm / BM);
  const dim3 gDF(Fm / BN, Mm / BM);
  for (int l = 0; l < Lm; ++l) {
    layernorm_k<<<Mm, 256, 0, stream>>>(h, ln1_a + l * Dm, ln1_b + l * Dm, xn);
    wmma_gemm<<<gDD, 256, 0, stream>>>(xn, Wq + (size_t)l * Dm * Dm, nullptr, nullptr, q, Mm, Dm, Dm, 0, 0);
    wmma_gemm<<<gDD, 256, 0, stream>>>(xn, Wk + (size_t)l * Dm * Dm, nullptr, nullptr, k, Mm, Dm, Dm, 0, 0);
    wmma_gemm<<<gDD, 256, 0, stream>>>(xn, Wv + (size_t)l * Dm * Dm, nullptr, nullptr, v, Mm, Dm, Dm, 0, 0);
    flash_attn<<<Bm * 12 * (Sm / 16), 32, 0, stream>>>(q, k, v, o);
    wmma_gemm<<<gDD, 256, 0, stream>>>(o, Wo + (size_t)l * Dm * Dm, nullptr, h, h, Mm, Dm, Dm, 0, 0);
    layernorm_k<<<Mm, 256, 0, stream>>>(h, ln2_a + l * Dm, ln2_b + l * Dm, xn);
    wmma_gemm<<<gDF, 256, 0, stream>>>(xn, W1 + (size_t)l * Dm * Fm, b1 + (size_t)l * Fm, nullptr, f, Mm, Fm, Dm, 0, 1);
    wmma_gemm<<<gDD, 256, 0, stream>>>(f, W2 + (size_t)l * Fm * Dm, b2 + (size_t)l * Dm, h, h, Mm, Dm, Fm, 0, 0);
  }
  layernorm_k<<<Mm, 256, 0, stream>>>(h, fn_a, fn_b, xn);
  const dim3 gV((Vm + BN - 1) / BN, Mm / BM);
  wmma_gemm<<<gV, 256, 0, stream>>>(xn, emb, pb, nullptr, out, Mm, Vm, Dm, 1, 0);
}
